// MSFIN_teacher_3556232921493
// MI455X (gfx1250) — matrix operand order repaired and checked by reading, not compiled, not run on hardware
//
#include <hip/hip_runtime.h>
#include <hip/hip_bf16.h>

typedef __attribute__((ext_vector_type(16))) _Float16 v16h;
typedef __attribute__((ext_vector_type(8)))  float    v8f;

struct U4 { unsigned long long x, y, z, w; };   // 32B carrier for v16h <-> LDS

// ---------------------------------------------------------------------------
// Fragment layouts per CDNA5 ISA 7.12.2 (wave32), f16 16x16x32:
//   A (16x32): row M = lane&15; this code: K = 0-15 (half 0) / 16-31 (half 1), contiguous
//   B (32x16): col N = lane&15; same contiguous K order as A (this code's choice, not the ISA table order)
//   C/D (16x16 f32, 8 VGPRs): VGPR r -> M = r + 8*half, N = lane&15
// ---------------------------------------------------------------------------
__device__ __forceinline__ v16h load_a_frag(const float* __restrict__ Abase,
                                            int k0, int half, int HW)
{
    v16h af;
    const float* ap0 = Abase + (size_t)(k0 + 16 * half) * HW;  // K group j=0..7
    const float* ap1 = ap0 + (size_t)8 * HW;                   // K group j=8..15
#pragma unroll
    for (int j = 0; j < 8; ++j) {
        af[j]     = (_Float16)ap0[(size_t)j * HW];
        af[j + 8] = (_Float16)ap1[(size_t)j * HW];
    }
    return af;
}

// ---------------------------------------------------------------------------
// Fused 1x1-conv GEMM on the CDNA5 WMMA path.
//   out[b, n, hw] = relu6?( gamma[n] * (sum_k A[b,k,hw] * W[n,k]) + beta[n] )
//                   (+ resid[b,n,hw] when resid != nullptr, applied post-affine)
// M = B*HW (always a multiple of 16 here), N = Cout, K = Cin.
//
// Tiling: block = 4 waves; each wave owns one 16-row M-tile and computes a
// 16x64 D tile (4 accumulators). Per K-step the 64x32 B-panel (shared by all
// 4 waves) is converted to f16 and staged in LDS ONCE per block (4x fewer
// global weight loads/converts), then each wave reads its fragments with
// ds_load_b64. Row stride padded to 36 halves (18 banks) to avoid conflicts.
// The strided A fragment is loaded from global once per K-step and reused for
// 4 WMMAs. All bounds handling is branchless (clamped addresses + selects);
// waves past the last M-tile are clamped onto it (duplicate identical stores)
// so every thread reaches the block barriers.
// ---------------------------------------------------------------------------
__global__ __launch_bounds__(128) void gemm1x1_wmma(
    const float* __restrict__ A,      // (B, K, HW)
    const float* __restrict__ W,      // (N, K)
    const float* __restrict__ gamma,  // (N) or nullptr -> 1.0
    const float* __restrict__ beta,   // (N) or nullptr -> 0.0
    const float* __restrict__ resid,  // (B, N, HW) or nullptr
    float* __restrict__ out,          // (B, N, HW)
    int Bn, int K, int N, int HW, int relu6)
{
    __shared__ unsigned long long Bs[64 * 9];   // 64 rows x 36 halves (72B stride)

    const int tid  = threadIdx.x;
    const int lane = tid & 31;
    const int wv   = tid >> 5;              // wave id in block (uniform)
    const int half = lane >> 4;             // 0: lanes 0-15, 1: lanes 16-31
    const int l16  = lane & 15;
    const int M    = Bn * HW;
    const int Mt   = M >> 4;                // # of 16-row M-tiles (M % 16 == 0)

    int mtile = blockIdx.x * 4 + wv;
    if (mtile >= Mt) mtile = Mt - 1;        // clamp: keep all waves in barriers
    const int ntile0 = blockIdx.y * 64;     // 4 N-subtiles of 16

    // Per-lane A row
    const int  m    = mtile * 16 + l16;
    const int  bimg = m / HW;
    const int  hw   = m % HW;
    const float* Abase = A + (size_t)bimg * K * HW + hw;

    // Staging role: thread -> (row, 16-col chunk) of the 64x32 B panel
    const int srow = tid >> 1;              // 0..63
    const int sks  = (tid & 1) * 16;        // 0 or 16
    const int nrow = ntile0 + srow;
    const float* Wrow = W + (size_t)(nrow < N ? nrow : N - 1) * K;

    v8f acc[4] = {};
    const int Kmain = K & ~31;
    int k0 = 0;
    for (; k0 < Kmain; k0 += 32) {
        // ---- cooperative stage: 64x32 f32 panel -> f16 in LDS -------------
        {
            v16h hv;
            const float4* sp = (const float4*)(Wrow + k0 + sks);
#pragma unroll
            for (int q = 0; q < 4; ++q) {
                float4 w4 = sp[q];
                hv[4 * q + 0] = (_Float16)w4.x;
                hv[4 * q + 1] = (_Float16)w4.y;
                hv[4 * q + 2] = (_Float16)w4.z;
                hv[4 * q + 3] = (_Float16)w4.w;
            }
            U4 u = __builtin_bit_cast(U4, hv);
            unsigned long long* dst = &Bs[srow * 9 + (tid & 1) * 4];
            dst[0] = u.x; dst[1] = u.y; dst[2] = u.z; dst[3] = u.w;
        }
        if (k0 + 32 < Kmain)
            __builtin_prefetch(Abase + (size_t)(k0 + 32 + 8 * half) * HW, 0, 1);
        __syncthreads();

        // ---- consume: A from global (once), B fragments from LDS ----------
        v16h af = load_a_frag(Abase, k0, half, HW);
#pragma unroll
        for (int q = 0; q < 4; ++q) {
            const unsigned long long* bsrc = &Bs[(q * 16 + l16) * 9 + half * 4];
            U4 bu;
            bu.x = bsrc[0]; bu.y = bsrc[1]; bu.z = bsrc[2]; bu.w = bsrc[3];
            v16h bf = __builtin_bit_cast(v16h, bu);
            acc[q] = __builtin_amdgcn_wmma_f32_16x16x32_f16(
                false, af, false, bf, (short)0, acc[q], false, false);
        }
        __syncthreads();
    }

    // K tail (K % 32 ∈ {16, 24}): direct global, clamped + selects, no barriers
    if (k0 < K) {
        v16h af;
#pragma unroll
        for (int j = 0; j < 16; ++j) {
            int ka = k0 + j + 16 * half;
            float av = Abase[(size_t)(ka < K ? ka : 0) * HW];
            af[j] = (_Float16)(ka < K ? av : 0.0f);
        }
#pragma unroll
        for (int q = 0; q < 4; ++q) {
            int nq = ntile0 + q * 16 + l16;
            const float* Wq = W + (size_t)(nq < N ? nq : N - 1) * K;
            v16h bf;
#pragma unroll
            for (int j = 0; j < 16; ++j) {
                int kb = k0 + j + 16 * half;
                float bv = Wq[kb < K ? kb : 0];
                bf[j] = (_Float16)(kb < K ? bv : 0.0f);
            }
            acc[q] = __builtin_amdgcn_wmma_f32_16x16x32_f16(
                false, af, false, bf, (short)0, acc[q], false, false);
        }
    }

    // Epilogue: rows of one lane are consecutive in m -> one div/mod, then a
    // branchless carry (HW >= 49 > 7, so po can wrap at most once).
    const int mbase = mtile * 16 + 8 * half;
    const int bo0   = mbase / HW;
    const int po0   = mbase % HW;
#pragma unroll
    for (int q = 0; q < 4; ++q) {
        int n = ntile0 + q * 16 + l16;
        if (n >= N) continue;                 // post-WMMA divergence is fine
        const float gsc = gamma ? gamma[n] : 1.0f;
        const float bof = beta  ? beta[n]  : 0.0f;
#pragma unroll
        for (int r = 0; r < 8; ++r) {
            int  po = po0 + r;
            bool wrap = po >= HW;
            int  bo = bo0 + (wrap ? 1 : 0);
            po = wrap ? po - HW : po;
            float v = acc[q][r] * gsc + bof;
            if (relu6) v = fminf(fmaxf(v, 0.0f), 6.0f);
            size_t oidx = ((size_t)bo * N + n) * HW + po;
            if (resid) v += resid[oidx];
            out[oidx] = v;
        }
    }
}

// ---------------------------------------------------------------------------
// Depthwise 3x3, pad 1, stride 1 or 2, fused BN + ReLU6. Bandwidth-bound VALU.
// Branchless taps: clamped index + mask select -> unconditional loads.
// ---------------------------------------------------------------------------
__global__ void dwconv3x3(const float* __restrict__ in, const float* __restrict__ w,
                          const float* __restrict__ g, const float* __restrict__ b,
                          float* __restrict__ out,
                          int Bn, int C, int Hin, int Hout, int stride)
{
    int idx = blockIdx.x * blockDim.x + threadIdx.x;
    int total = Bn * C * Hout * Hout;
    if (idx >= total) return;
    int wo = idx % Hout;
    int ho = (idx / Hout) % Hout;
    int c  = (idx / (Hout * Hout)) % C;
    int bi = idx / (Hout * Hout * C);

    const float* ib = in + ((size_t)bi * C + c) * Hin * Hin;
    const float* wc = w + c * 9;
    float acc = 0.0f;
#pragma unroll
    for (int kh = 0; kh < 3; ++kh) {
        int  hi = ho * stride - 1 + kh;
        bool hv = (unsigned)hi < (unsigned)Hin;
        int  hic = hv ? hi : 0;
#pragma unroll
        for (int kw = 0; kw < 3; ++kw) {
            int  wi = wo * stride - 1 + kw;
            bool wv = (unsigned)wi < (unsigned)Hin;
            int  wic = wv ? wi : 0;
            float v = ib[hic * Hin + wic];
            acc += ((hv && wv) ? v : 0.0f) * wc[kh * 3 + kw];
        }
    }
    float v = acc * g[c] + b[c];
    out[idx] = fminf(fmaxf(v, 0.0f), 6.0f);
}

// ---------------------------------------------------------------------------
// Stem: conv 3x3, 3->32, stride 2, pad 1, fused BN + ReLU6. 224 -> 112.
// ---------------------------------------------------------------------------
__global__ void stem_conv(const float* __restrict__ x, const float* __restrict__ w,
                          const float* __restrict__ g, const float* __restrict__ b,
                          float* __restrict__ out, int Bn)
{
    const int HO = 112, HI = 224, HW = HO * HO;
    int idx = blockIdx.x * blockDim.x + threadIdx.x;
    int total = Bn * 32 * HW;
    if (idx >= total) return;
    int hw = idx % HW;
    int c  = (idx / HW) % 32;
    int bi = idx / (HW * 32);
    int ho = hw / HO, wo = hw % HO;

    float acc = 0.0f;
#pragma unroll
    for (int ci = 0; ci < 3; ++ci) {
        const float* ib = x + ((size_t)bi * 3 + ci) * HI * HI;
        const float* wc = w + (c * 3 + ci) * 9;
#pragma unroll
        for (int kh = 0; kh < 3; ++kh) {
            int  hi = ho * 2 - 1 + kh;
            bool hv = (unsigned)hi < (unsigned)HI;
            int  hic = hv ? hi : 0;
#pragma unroll
            for (int kw = 0; kw < 3; ++kw) {
                int  wi = wo * 2 - 1 + kw;
                bool wv = (unsigned)wi < (unsigned)HI;
                int  wic = wv ? wi : 0;
                float v = ib[hic * HI + wic];
                acc += ((hv && wv) ? v : 0.0f) * wc[kh * 3 + kw];
            }
        }
    }
    float v = acc * g[c] + b[c];
    out[idx] = fminf(fmaxf(v, 0.0f), 6.0f);
}

// ---------------------------------------------------------------------------
// Host orchestration
// ---------------------------------------------------------------------------
static inline void launch_gemm(const float* A, const float* W, const float* g,
                               const float* b, const float* resid, float* out,
                               int Bn, int K, int N, int HW, int relu6,
                               hipStream_t stream)
{
    int M = Bn * HW;
    dim3 grid((M + 63) / 64, (N + 63) / 64);   // 4 M-tiles/block, 64 N/block
    gemm1x1_wmma<<<grid, 128, 0, stream>>>(A, W, g, b, resid, out, Bn, K, N, HW, relu6);
}

extern "C" void kernel_launch(void* const* d_in, const int* in_sizes, int n_in,
                              void* d_out, int out_size, void* d_ws, size_t ws_size,
                              hipStream_t stream)
{
    (void)n_in; (void)out_size; (void)ws_size;

    // MobileNetV2 config: (expand t, out channels c, repeats n, stride s)
    const int cfg_t[7] = {1, 6, 6, 6, 6, 6, 6};
    const int cfg_c[7] = {16, 24, 32, 64, 96, 160, 320};
    const int cfg_n[7] = {1, 2, 3, 4, 3, 3, 1};
    const int cfg_s[7] = {1, 2, 2, 2, 1, 2, 1};

    // Input pointer walk: setup_inputs() dict insertion order.
    int p = 0;
    const float* x      = (const float*)d_in[p++];
    const int    Bn     = in_sizes[0] / (3 * 224 * 224);   // batch (=16)
    const float* stem_w = (const float*)d_in[p++];
    const float* stem_g = (const float*)d_in[p++];
    const float* stem_b = (const float*)d_in[p++];

    // Workspace layout (floats). Caps sized to the largest tensors seen:
    //   X/P (block in/out): 16*32*112*112       = 6,422,528
    //   E   (expand out):   16*96*112*112       = 19,267,584
    //   D   (dw out):       16*144*56*56        = 7,225,344
    float* bufX = (float*)d_ws;
    float* bufE = bufX + 6422528;
    float* bufD = bufE + 19267584;
    float* bufP = bufD + 7225344;

    // Stem: (B,3,224,224) -> (B,32,112,112)
    {
        int tot = Bn * 32 * 112 * 112;
        stem_conv<<<(tot + 255) / 256, 256, 0, stream>>>(x, stem_w, stem_g, stem_b, bufX, Bn);
    }

    float* cur = bufX;
    float* nxt = bufP;
    int H = 112, cin = 32;

    for (int s7 = 0; s7 < 7; ++s7) {
        int t = cfg_t[s7], c = cfg_c[s7], reps = cfg_n[s7], st0 = cfg_s[s7];
        for (int i = 0; i < reps; ++i) {
            const float *ew = nullptr, *eg = nullptr, *eb = nullptr;
            if (t != 1) {
                ew = (const float*)d_in[p++];
                eg = (const float*)d_in[p++];
                eb = (const float*)d_in[p++];
            }
            const float* dwW = (const float*)d_in[p++];
            const float* dgg = (const float*)d_in[p++];
            const float* dbb = (const float*)d_in[p++];
            const float* pw  = (const float*)d_in[p++];
            const float* pg  = (const float*)d_in[p++];
            const float* pb  = (const float*)d_in[p++];

            int stride = (i == 0) ? st0 : 1;
            int hid    = cin * t;
            int Hout   = (stride == 2) ? H / 2 : H;
            bool res   = (stride == 1 && cin == c);

            const float* din = cur;
            if (t != 1) {   // expand 1x1 (WMMA) + BN + ReLU6
                launch_gemm(cur, ew, eg, eb, nullptr, bufE, Bn, cin, hid, H * H, 1, stream);
                din = bufE;
            }
            {               // depthwise 3x3 + BN + ReLU6
                int tot = Bn * hid * Hout * Hout;
                dwconv3x3<<<(tot + 255) / 256, 256, 0, stream>>>(
                    din, dwW, dgg, dbb, bufD, Bn, hid, H, Hout, stride);
            }
            // project 1x1 (WMMA) + BN, linear bottleneck, optional residual
            launch_gemm(bufD, pw, pg, pb, res ? cur : nullptr, nxt,
                        Bn, hid, c, Hout * Hout, 0, stream);

            float* tmp = cur; cur = nxt; nxt = tmp;
            H = Hout; cin = c;
        }
    }

    const float* head_w = (const float*)d_in[p++];
    const float* head_g = (const float*)d_in[p++];
    const float* head_b = (const float*)d_in[p++];
    const float* fc_w   = (const float*)d_in[p++];
    const float* fc_b   = (const float*)d_in[p++];

    // Head 1x1: 320 -> 1280 @7x7 (WMMA) + BN + ReLU6
    launch_gemm(cur, head_w, head_g, head_b, nullptr, nxt, Bn, 320, 1280, 49, 1, stream);
    { float* tmp = cur; cur = nxt; nxt = tmp; }

    // CAM: einsum('bchw,nc->bnhw') + fc_b  == GEMM M=B*49, N=1000, K=1280 (WMMA)
    launch_gemm(cur, fc_w, nullptr, fc_b, nullptr, (float*)d_out, Bn, 1280, 1000, 49, 0, stream);
}
